// TokenSelector_33638183862966
// MI455X (gfx1250) — compile-verified, matrix operand order checked by reading, not run on hardware
//
#include <hip/hip_runtime.h>
#include <hip/hip_bf16.h>

// ---------------- problem constants (from reference) ----------------
static constexpr int Bc  = 2;
static constexpr int Hc  = 8;
static constexpr int HPc = 16;
static constexpr int Tc  = 1024;
static constexpr int Ec  = 512;
static constexpr int HDc = 64;
static constexpr float SCALEc = 0.125f;   // HD^-0.5

typedef __attribute__((ext_vector_type(16))) __bf16 v16bf;
typedef __attribute__((ext_vector_type(8)))  __bf16 v8bf;
typedef __attribute__((ext_vector_type(8)))  float  v8f;
typedef __attribute__((ext_vector_type(4)))  int    v4i;

// Pointer types matching the async-to-LDS builtin prototype:
// (v4i addrspace(1)*, v4i addrspace(3)*, imm offset, imm cpol)
typedef __attribute__((address_space(1))) v4i* as1_v4i_ptr;
typedef __attribute__((address_space(3))) v4i* as3_v4i_ptr;

#if defined(__HIP_DEVICE_COMPILE__) && defined(__has_builtin)
#if __has_builtin(__builtin_amdgcn_global_load_async_to_lds_b128)
#define HAVE_ASYNC_LDS 1
#endif
#if __has_builtin(__builtin_amdgcn_s_wait_asynccnt)
#define HAVE_WAIT_ASYNC 1
#endif
#endif

// =====================================================================
// Kernel 1: 1x1 expand conv  A[B,H,T,T] -> A_expanded[B,HP,T,T]
// =====================================================================
__global__ __launch_bounds__(256) void expand_kernel(
    const float* __restrict__ A, const float* __restrict__ We,
    float* __restrict__ Aexp)
{
    __shared__ float we[HPc * Hc];
    const int tid = threadIdx.x;
    if (tid < HPc * Hc) we[tid] = We[tid];
    __syncthreads();

    const long g  = (long)blockIdx.x * 256 + tid;   // 0 .. B*T*T
    const int  b  = (int)(g >> 20);
    const int  xy = (int)(g & ((1 << 20) - 1));

    float a[Hc];
#pragma unroll
    for (int h = 0; h < Hc; ++h)
        a[h] = A[((long)(b * Hc + h) << 20) + xy];

#pragma unroll
    for (int p = 0; p < HPc; ++p) {
        float s = 0.f;
#pragma unroll
        for (int h = 0; h < Hc; ++h) s = fmaf(we[p * Hc + h], a[h], s);
        Aexp[((long)(b * HPc + p) << 20) + xy] = s;
    }
}

// =====================================================================
// Kernel 2: fused depthwise 3x3 + bias + ReLU + 1x1 reduce + scale +
//           row softmax.  One block per (b, x).  Writes bf16 probs.
// =====================================================================
__global__ __launch_bounds__(256) void conv_reduce_softmax_kernel(
    const float* __restrict__ Aexp, const float* __restrict__ Wl,
    const float* __restrict__ Bl,  const float* __restrict__ Wr,
    __bf16* __restrict__ attn)
{
    __shared__ float tileS[HPc][3][260];
    __shared__ float wl[HPc * 9];
    __shared__ float wr[Hc * HPc];
    __shared__ float bl[HPc];
    __shared__ float red[256];

    const int tid = threadIdx.x;
    const int b   = blockIdx.x >> 10;
    const int x   = blockIdx.x & (Tc - 1);

    if (tid < HPc * 9)  wl[tid] = Wl[tid];
    if (tid < Hc * HPc) wr[tid] = Wr[tid];
    if (tid < HPc)      bl[tid] = Bl[tid];

    float logits[4][Hc];
#pragma unroll
    for (int t = 0; t < 4; ++t)
#pragma unroll
        for (int h = 0; h < Hc; ++h) logits[t][h] = 0.f;

    for (int t = 0; t < 4; ++t) {
        __syncthreads();
        for (int p = 0; p < HPc; ++p) {
            for (int dx = 0; dx < 3; ++dx) {
                const int gx = x + dx - 1;
                const float* src =
                    Aexp + ((long)(b * HPc + p) << 20) + (long)gx * Tc;
                for (int yy = tid; yy < 258; yy += 256) {
                    const int gy = t * 256 + yy - 1;
                    float v = 0.f;
                    if (gx >= 0 && gx < Tc && gy >= 0 && gy < Tc) v = src[gy];
                    tileS[p][dx][yy] = v;
                }
            }
        }
        __syncthreads();
#pragma unroll
        for (int p = 0; p < HPc; ++p) {
            float c = bl[p];
#pragma unroll
            for (int dx = 0; dx < 3; ++dx)
#pragma unroll
                for (int dy = 0; dy < 3; ++dy)
                    c = fmaf(wl[p * 9 + dx * 3 + dy], tileS[p][dx][tid + dy], c);
            const float v = fmaxf(c, 0.f);
#pragma unroll
            for (int h = 0; h < Hc; ++h)
                logits[t][h] = fmaf(wr[h * HPc + p], v, logits[t][h]);
        }
    }

    for (int h = 0; h < Hc; ++h) {
        float m = -3.4e38f;
#pragma unroll
        for (int t = 0; t < 4; ++t) m = fmaxf(m, logits[t][h] * SCALEc);
        __syncthreads();
        red[tid] = m;
        __syncthreads();
        for (int s = 128; s > 0; s >>= 1) {
            if (tid < s) red[tid] = fmaxf(red[tid], red[tid + s]);
            __syncthreads();
        }
        const float rowmax = red[0];

        float e[4], ssum = 0.f;
#pragma unroll
        for (int t = 0; t < 4; ++t) {
            e[t] = __expf(logits[t][h] * SCALEc - rowmax);
            ssum += e[t];
        }
        __syncthreads();
        red[tid] = ssum;
        __syncthreads();
        for (int s = 128; s > 0; s >>= 1) {
            if (tid < s) red[tid] += red[tid + s];
            __syncthreads();
        }
        const float inv = 1.0f / red[0];

        const long base = ((long)(b * Hc + h) * Tc + x) * (long)Tc;
#pragma unroll
        for (int t = 0; t < 4; ++t)
            attn[base + t * 256 + tid] = (__bf16)(e[t] * inv);
    }
}

// =====================================================================
// WMMA fragment helper.  Tiles stored "fragment-row-major":
//   A tile: [m][k], B tile: [n][k] (B^T); LDT=40 bf16 -> 80B rows,
//   16B-aligned chunks -> 2x ds_load_b128 per fragment.
// ISA 16-bit operand layout: lanes 0-15 hold K 0..7 / 16..23,
// lanes 16-31 hold K 8..15 / 24..31.
// =====================================================================
static constexpr int LDT = 40;

__device__ __forceinline__ v16bf load_frag(const __bf16* base, int idx0, int lane)
{
    const int r  = lane & 15;
    const int hi = (lane >> 4) & 1;
    const __bf16* p = base + (idx0 + r) * LDT;
    const v8bf lo = *(const v8bf*)(p + hi * 8);
    const v8bf hh = *(const v8bf*)(p + 16 + hi * 8);
    v16bf out;
#pragma unroll
    for (int e = 0; e < 8; ++e) { out[e] = lo[e]; out[e + 8] = hh[e]; }
    return out;
}

// ---- staging helpers for the attn@V kernel ----
__device__ __forceinline__ void stage_attn(const __bf16* Ag, int k0,
                                           __bf16* dst, int tid)
{
#if defined(HAVE_ASYNC_LDS)
#pragma unroll
    for (int i = 0; i < 2; ++i) {
        const int idx = i * 256 + tid;          // 512 x 16B chunks
        const int r = idx >> 2, c = idx & 3;
        const char* g = (const char*)Ag + (size_t)r * (Tc * 2) +
                        (size_t)k0 * 2 + c * 16;
        char* l = (char*)dst + (r * LDT + c * 8) * 2;
        __builtin_amdgcn_global_load_async_to_lds_b128(
            (as1_v4i_ptr)(void*)g, (as3_v4i_ptr)(void*)l, 0, 0);
    }
#else
    unsigned* du = (unsigned*)dst;
    const unsigned* Agu = (const unsigned*)Ag;
#pragma unroll
    for (int i = 0; i < 8; ++i) {
        const int idx = i * 256 + tid;
        const int r = idx >> 4, c = idx & 15;
        du[r * (LDT / 2) + c] = Agu[r * 512 + (k0 >> 1) + c];
    }
#endif
}

__device__ __forceinline__ void stage_v(const float* Vg, int k0,
                                        __bf16* dst, int tid)
{
#pragma unroll
    for (int i = 0; i < 8; ++i) {
        const int idx = i * 256 + tid;
        const int r = idx >> 6, c = idx & 63;   // r = k local, c = n
        dst[c * LDT + r] = (__bf16)Vg[(k0 + r) * HDc + c];
    }
}

template <int N>
__device__ __forceinline__ void wait_async()
{
#if defined(HAVE_ASYNC_LDS)
#if defined(HAVE_WAIT_ASYNC)
    __builtin_amdgcn_s_wait_asynccnt(N);
#else
    if (N == 0) asm volatile("s_wait_asynccnt 0x0" ::: "memory");
    else        asm volatile("s_wait_asynccnt 0x2" ::: "memory");
#endif
#endif
}

// =====================================================================
// Kernel 3: attn @ V  (per (b,h): 1024 x 64 x 1024 GEMM, bf16 WMMA)
// Double-buffered: async-to-LDS fetch of tile k+1 overlaps WMMA of
// tile k; s_wait_asynccnt 2 leaves the in-flight tile outstanding.
// =====================================================================
__global__ __launch_bounds__(256) void attnv_wmma_kernel(
    const __bf16* __restrict__ attn, const float* __restrict__ V,
    float* __restrict__ outw)
{
    __shared__ alignas(16) __bf16 As[2][128 * LDT];   // attn tiles [m][k]
    __shared__ alignas(16) __bf16 Bs[2][64 * LDT];    // V^T tiles  [n][k]

    const int tid  = threadIdx.x;
    const int lane = tid & 31;
    const int wave = tid >> 5;
    const int bh   = blockIdx.x >> 3;
    const int mblk = blockIdx.x & 7;
    const int b    = bh >> 3;
    const int h    = bh & 7;

    const __bf16* Ag = attn + ((long)bh << 20) + (long)mblk * 128 * Tc;
    const float* Vg  = V + (long)bh * Tc * HDc;

    constexpr int NK = Tc / 32;   // 32 K-tiles

    // prologue: stage tile 0
    stage_attn(Ag, 0, As[0], tid);
    stage_v(Vg, 0, Bs[0], tid);

    v8f acc[4] = {{}, {}, {}, {}};

    for (int kt = 0; kt < NK; ++kt) {
        const int cur = kt & 1;
        const int nxt = cur ^ 1;

        if (kt + 1 < NK) {
            // issue next tile while current is still in flight / computing
            stage_attn(Ag, (kt + 1) * 32, As[nxt], tid);
            stage_v(Vg, (kt + 1) * 32, Bs[nxt], tid);
            wait_async<2>();          // tile kt resident, tile kt+1 in flight
        } else {
            wait_async<0>();
        }
        __syncthreads();              // LDS writes for tile kt visible to all

        const v16bf a = load_frag(As[cur], wave * 16, lane);
#pragma unroll
        for (int nt = 0; nt < 4; ++nt) {
            const v16bf bb = load_frag(Bs[cur], nt * 16, lane);
            acc[nt] = __builtin_amdgcn_wmma_f32_16x16x32_bf16(
                false, a, false, bb, (short)0, acc[nt], false, false);
        }
        __syncthreads();              // reads done before nxt buffer reuse
    }

    // epilogue: C VGPR e -> M = e + 8*(lane>=16), N = lane&15
    const int r     = lane & 15;
    const int hi    = lane >> 4;
    const int mbase = mblk * 128 + wave * 16 + hi * 8;
#pragma unroll
    for (int nt = 0; nt < 4; ++nt) {
#pragma unroll
        for (int e = 0; e < 8; ++e) {
            const int m   = mbase + e;
            const int col = h * HDc + nt * 16 + r;
            outw[((long)b * Tc + m) * Ec + col] = acc[nt][e];
        }
    }
}

// =====================================================================
// Kernel 4: projection  out = X @ W_proj^T + b_proj  (2048 x 512 x 512)
// =====================================================================
__global__ __launch_bounds__(256) void proj_wmma_kernel(
    const float* __restrict__ X, const float* __restrict__ W,
    const float* __restrict__ bias, float* __restrict__ out)
{
    __shared__ alignas(16) __bf16 As[128 * LDT];   // X tile  [m][k]
    __shared__ alignas(16) __bf16 Bs[64 * LDT];    // W tile  [n][k]

    const int tid  = threadIdx.x;
    const int lane = tid & 31;
    const int wave = tid >> 5;
    const int mblk = blockIdx.x >> 3;
    const int nblk = blockIdx.x & 7;

    const float* Xg = X + (long)mblk * 128 * Ec;

    v8f acc[4] = {{}, {}, {}, {}};

    for (int k0 = 0; k0 < Ec; k0 += 32) {
#pragma unroll
        for (int i = 0; i < 16; ++i) {
            const int idx = i * 256 + tid;
            const int r = idx >> 5, c = idx & 31;
            As[r * LDT + c] = (__bf16)Xg[r * Ec + k0 + c];
        }
#pragma unroll
        for (int i = 0; i < 8; ++i) {
            const int idx = i * 256 + tid;
            const int n = idx >> 5, c = idx & 31;
            Bs[n * LDT + c] = (__bf16)W[(nblk * 64 + n) * Ec + k0 + c];
        }
        __syncthreads();

        const v16bf a = load_frag(As, wave * 16, lane);
#pragma unroll
        for (int nt = 0; nt < 4; ++nt) {
            const v16bf bb = load_frag(Bs, nt * 16, lane);
            acc[nt] = __builtin_amdgcn_wmma_f32_16x16x32_bf16(
                false, a, false, bb, (short)0, acc[nt], false, false);
        }
        __syncthreads();
    }

    const int r     = lane & 15;
    const int hi    = lane >> 4;
    const int mbase = mblk * 128 + wave * 16 + hi * 8;
#pragma unroll
    for (int nt = 0; nt < 4; ++nt) {
#pragma unroll
        for (int e = 0; e < 8; ++e) {
            const int m = mbase + e;
            const int n = nblk * 64 + nt * 16 + r;
            out[(long)m * Ec + n] = acc[nt][e] + bias[n];
        }
    }
}

// =====================================================================
// Launcher
// =====================================================================
extern "C" void kernel_launch(void* const* d_in, const int* in_sizes, int n_in,
                              void* d_out, int out_size, void* d_ws, size_t ws_size,
                              hipStream_t stream)
{
    const float* A        = (const float*)d_in[0];
    // d_in[1] = Q, d_in[2] = K : unused by the reference
    const float* V        = (const float*)d_in[3];
    const float* W_expand = (const float*)d_in[4];
    const float* W_local  = (const float*)d_in[5];
    const float* b_local  = (const float*)d_in[6];
    const float* W_reduce = (const float*)d_in[7];
    const float* W_proj   = (const float*)d_in[8];
    const float* b_proj   = (const float*)d_in[9];

    float* out  = (float*)d_out;                       // [B,T,E]
    float* Aexp = out + (long)Bc * Tc * Ec;            // [B,HP,T,T] output slot

    __bf16* attn = (__bf16*)d_ws;                      // 33.5 MB bf16 probs
    float* outw  = (float*)((char*)d_ws +
                            (size_t)Bc * Hc * Tc * Tc * sizeof(__bf16));

    expand_kernel<<<(Bc * Tc * Tc) / 256, 256, 0, stream>>>(A, W_expand, Aexp);

    conv_reduce_softmax_kernel<<<Bc * Tc, 256, 0, stream>>>(
        Aexp, W_local, b_local, W_reduce, attn);

    attnv_wmma_kernel<<<Bc * Hc * (Tc / 128), 256, 0, stream>>>(attn, V, outw);

    proj_wmma_kernel<<<(Bc * Tc / 128) * (Ec / 64), 256, 0, stream>>>(
        outw, W_proj, b_proj, out);
}
